// ECPEC_74672301408830
// MI455X (gfx1250) — matrix operand order repaired and checked by reading, not compiled, not run on hardware
//
#include <hip/hip_runtime.h>
#include <hip/hip_bf16.h>
#include <math.h>

// ---------------------------------------------------------------------------
// ECPE model on MI455X (gfx1250): all GEMMs via v_wmma_f32_16x16x32_bf16.
//  - wmma_gemm_2x2: register-blocked (32x32 per wave) for the big GEMMs
//  - wmma_gemm    : 16x16 per wave for small/latency-bound GEMMs (LSTM steps)
// ---------------------------------------------------------------------------

typedef unsigned short u16;
typedef __attribute__((ext_vector_type(16))) __bf16 v16bf;
typedef __attribute__((ext_vector_type(8)))  float  v8f;

#define B_   256
#define T_   96
#define D_   768
#define H_   384
#define G4_  1536
#define NE_  128
#define NC_  256
#define NK_  64
#define POS_ 50
#define KP_  2368   // 2355/2356 features padded up to a multiple of 32

__device__ __forceinline__ u16 f2bf(float x) {
  unsigned u = __float_as_uint(x);
  u += 0x7FFFu + ((u >> 16) & 1u);          // round-to-nearest-even
  return (u16)(u >> 16);
}
__device__ __forceinline__ float sigf(float x) { return 1.f / (1.f + expf(-x)); }
__device__ __forceinline__ float lrelu(float x) { return x > 0.f ? x : 0.01f * x; }

union FragU { v16bf v; uint4 q[2]; };

// ---------------------------------------------------------------------------
// 1x1 WMMA GEMM:  C[M,N](f32) = act( A[M,K]bf16 @ B[N,K]^T bf16 + bias + Cin )
// M%32==0, N%64==0, K%32==0, lda/ldb%8==0. Block=8 waves, tile 32x64.
// ---------------------------------------------------------------------------
__global__ __launch_bounds__(256) void wmma_gemm(
    const u16* __restrict__ A, int lda,
    const u16* __restrict__ B, int ldb,
    const float* __restrict__ bias,
    const float* __restrict__ Cin, long long ldcin,
    float* __restrict__ C, int ldc,
    int M, int N, int K, int act)
{
  const int lane = threadIdx.x & 31;
  const int wid  = threadIdx.x >> 5;
  const int mt = blockIdx.y * 32 + (wid >> 2) * 16;
  const int nt = blockIdx.x * 64 + (wid & 3) * 16;
  if (mt >= M || nt >= N) return;

  const int half = lane >> 4;
  const int l15  = lane & 15;

  // A 16x32 bf16 fragment: lanes 0-15 -> K {0..7,16..23}; 16-31 -> {8..15,24..31}
  const u16* arow = A + (size_t)(mt + l15) * lda + half * 8;
  // B 32x16 bf16 fragment: lanes 0-15 -> K {0..7,16..23}; 16-31 -> {8..15,24..31}
  const u16* brow = B + (size_t)(nt + l15) * ldb + half * 8;

  v8f acc = {0.f, 0.f, 0.f, 0.f, 0.f, 0.f, 0.f, 0.f};
  for (int k0 = 0; k0 < K; k0 += 32) {
    FragU a, b;
    a.q[0] = *reinterpret_cast<const uint4*>(arow + k0);
    a.q[1] = *reinterpret_cast<const uint4*>(arow + k0 + 16);
    b.q[0] = *reinterpret_cast<const uint4*>(brow + k0);
    b.q[1] = *reinterpret_cast<const uint4*>(brow + k0 + 16);
    acc = __builtin_amdgcn_wmma_f32_16x16x32_bf16(
        false, a.v, false, b.v, (short)0, acc, false, false);
  }

  const int cn = nt + l15;
  const float bv = bias ? bias[cn] : 0.f;
#pragma unroll
  for (int r = 0; r < 8; ++r) {
    const int cm = mt + r + half * 8;
    float v = acc[r] + bv;
    if (Cin) v += Cin[(size_t)cm * ldcin + cn];
    if (act) v = lrelu(v);
    C[(size_t)cm * ldc + cn] = v;
  }
}

// ---------------------------------------------------------------------------
// 2x2 register-blocked WMMA GEMM: 32x32 per wave, block tile 64x128.
// M%64==0, N%128==0, K%32==0. 8 b128 loads -> 4 WMMAs per k-step.
// ---------------------------------------------------------------------------
__global__ __launch_bounds__(256) void wmma_gemm_2x2(
    const u16* __restrict__ A, int lda,
    const u16* __restrict__ B, int ldb,
    const float* __restrict__ bias,
    float* __restrict__ C, int ldc,
    int M, int N, int K, int act)
{
  const int lane = threadIdx.x & 31;
  const int wid  = threadIdx.x >> 5;
  const int mt = blockIdx.y * 64 + (wid >> 2) * 32;
  const int nt = blockIdx.x * 128 + (wid & 3) * 32;
  if (mt >= M || nt >= N) return;

  const int half = lane >> 4;
  const int l15  = lane & 15;

  const u16* arow0 = A + (size_t)(mt + l15) * lda + half * 8;
  const u16* arow1 = arow0 + (size_t)16 * lda;
  const u16* brow0 = B + (size_t)(nt + l15) * ldb + half * 8;
  const u16* brow1 = brow0 + (size_t)16 * ldb;

  v8f acc00 = {0.f,0.f,0.f,0.f,0.f,0.f,0.f,0.f};
  v8f acc01 = acc00, acc10 = acc00, acc11 = acc00;

  for (int k0 = 0; k0 < K; k0 += 32) {
    FragU a0, a1, b0, b1;
    a0.q[0] = *reinterpret_cast<const uint4*>(arow0 + k0);
    a0.q[1] = *reinterpret_cast<const uint4*>(arow0 + k0 + 16);
    a1.q[0] = *reinterpret_cast<const uint4*>(arow1 + k0);
    a1.q[1] = *reinterpret_cast<const uint4*>(arow1 + k0 + 16);
    b0.q[0] = *reinterpret_cast<const uint4*>(brow0 + k0);
    b0.q[1] = *reinterpret_cast<const uint4*>(brow0 + k0 + 16);
    b1.q[0] = *reinterpret_cast<const uint4*>(brow1 + k0);
    b1.q[1] = *reinterpret_cast<const uint4*>(brow1 + k0 + 16);
    acc00 = __builtin_amdgcn_wmma_f32_16x16x32_bf16(
        false, a0.v, false, b0.v, (short)0, acc00, false, false);
    acc01 = __builtin_amdgcn_wmma_f32_16x16x32_bf16(
        false, a0.v, false, b1.v, (short)0, acc01, false, false);
    acc10 = __builtin_amdgcn_wmma_f32_16x16x32_bf16(
        false, a1.v, false, b0.v, (short)0, acc10, false, false);
    acc11 = __builtin_amdgcn_wmma_f32_16x16x32_bf16(
        false, a1.v, false, b1.v, (short)0, acc11, false, false);
  }

  const int cn0 = nt + l15, cn1 = nt + 16 + l15;
  const float bv0 = bias ? bias[cn0] : 0.f;
  const float bv1 = bias ? bias[cn1] : 0.f;
#pragma unroll
  for (int r = 0; r < 8; ++r) {
    const int cm0 = mt + r + half * 8;
    const int cm1 = cm0 + 16;
    float v;
    v = acc00[r] + bv0; if (act) v = lrelu(v); C[(size_t)cm0 * ldc + cn0] = v;
    v = acc01[r] + bv1; if (act) v = lrelu(v); C[(size_t)cm0 * ldc + cn1] = v;
    v = acc10[r] + bv0; if (act) v = lrelu(v); C[(size_t)cm1 * ldc + cn0] = v;
    v = acc11[r] + bv1; if (act) v = lrelu(v); C[(size_t)cm1 * ldc + cn1] = v;
  }
}

// ---------------------------------------------------------------------------
// Small helper kernels
// ---------------------------------------------------------------------------
__global__ void cvt_bf(const float* __restrict__ s, u16* __restrict__ d, size_t n) {
  size_t i = (size_t)blockIdx.x * blockDim.x + threadIdx.x;
  if (i < n) d[i] = f2bf(s[i]);
}

// dst[c*Rpad + r] = (r<R) ? src[r*C + c] : 0   (transpose [R,C] -> [C,Rpad] bf16)
__global__ void cvt_tr_pad(const float* __restrict__ s, u16* __restrict__ d,
                           int R, int Ccols, int Rpad) {
  size_t i = (size_t)blockIdx.x * blockDim.x + threadIdx.x;
  size_t tot = (size_t)Ccols * Rpad;
  if (i >= tot) return;
  int c = (int)(i / Rpad), r = (int)(i % Rpad);
  d[i] = (r < R) ? f2bf(s[(size_t)r * Ccols + c]) : (u16)0;
}

__global__ void add2(const float* a, const float* b, float* o, int n) {
  int i = blockIdx.x * blockDim.x + threadIdx.x;
  if (i < n) o[i] = a[i] + b[i];
}

// LSTM gate nonlinearity + state update; gt [B,4H] pre-activation gates.
__global__ void lstm_pw(const float* __restrict__ gt, float* __restrict__ cst,
                        float* __restrict__ hdst, size_t hstride,
                        u16* __restrict__ hbf) {
  int i = blockIdx.x * blockDim.x + threadIdx.x;
  if (i >= B_ * H_) return;
  int b = i / H_, j = i % H_;
  const float* g = gt + (size_t)b * G4_;
  float ig = sigf(g[j]);
  float fg = sigf(g[H_ + j]);
  float gg = tanhf(g[2 * H_ + j]);
  float og = sigf(g[3 * H_ + j]);
  float cv = fg * cst[i] + ig * gg;
  cst[i] = cv;
  float h = og * tanhf(cv);
  hdst[(size_t)b * hstride + j] = h;
  hbf[i] = f2bf(h);
}

// x_rev[b,t,:] = x[b, rev_idx(b,t), :]  (bf16 rows of width D_)
__global__ void gather_rev_bf(const u16* __restrict__ src, u16* __restrict__ dst,
                              const int* __restrict__ len) {
  int bt = blockIdx.x, b = bt / T_, t = bt % T_;
  int L = len[b];
  int rid = (t < L) ? (L - 1 - t) : t;
  const u16* s = src + ((size_t)b * T_ + rid) * D_;
  u16* d = dst + (size_t)bt * D_;
  for (int j = threadIdx.x; j < D_; j += blockDim.x) d[j] = s[j];
}

// Hs[b,t,384+j] = tmp[b, rev_idx(b,t), j]
__global__ void gather_rev_h(const float* __restrict__ tmp, float* __restrict__ Hs,
                             const int* __restrict__ len) {
  int bt = blockIdx.x, b = bt / T_, t = bt % T_;
  int L = len[b];
  int rid = (t < L) ? (L - 1 - t) : t;
  const float* s = tmp + ((size_t)b * T_ + rid) * H_;
  float* d = Hs + (size_t)bt * D_ + H_;
  for (int j = threadIdx.x; j < H_; j += blockDim.x) d[j] = s[j];
}

// e[row] = dot(S1[row,:], w)   (one block per row)
__global__ void rowdot(const float* __restrict__ S1, const float* __restrict__ w,
                       float* __restrict__ e, int D) {
  __shared__ float sh[256];
  size_t r = blockIdx.x;
  float acc = 0.f;
  for (int d = threadIdx.x; d < D; d += 256) acc += S1[r * D + d] * w[d];
  sh[threadIdx.x] = acc; __syncthreads();
  for (int s = 128; s > 0; s >>= 1) {
    if (threadIdx.x < s) sh[threadIdx.x] += sh[threadIdx.x + s];
    __syncthreads();
  }
  if (threadIdx.x == 0) e[r] = sh[0];
}

// masked softmax_by_length + weighted pooling: U[b,:] = sum_t alpha * Hs[b,t,:]
__global__ void attn_u(const float* __restrict__ e, const float* __restrict__ Hs,
                       const int* __restrict__ len, float* __restrict__ U,
                       u16* __restrict__ Ubf) {
  __shared__ float al[T_];
  __shared__ float red[256];
  int b = blockIdx.x, L = len[b];
  float v = 0.f;
  if (threadIdx.x < T_) {
    v = (threadIdx.x < L) ? e[b * T_ + threadIdx.x] : 0.f;
    al[threadIdx.x] = v;
  }
  red[threadIdx.x] = (threadIdx.x < T_) ? v : 0.f;
  __syncthreads();
  for (int s = 128; s > 0; s >>= 1) {
    if (threadIdx.x < s) red[threadIdx.x] += red[threadIdx.x + s];
    __syncthreads();
  }
  float inv = 1.f / (red[0] + 1e-9f);
  for (int d = threadIdx.x; d < D_; d += 256) {
    float acc = 0.f;
    for (int t = 0; t < T_; ++t) acc += al[t] * Hs[((size_t)b * T_ + t) * D_ + d];
    acc *= inv;
    U[(size_t)b * D_ + d] = acc;
    Ubf[(size_t)b * D_ + d] = f2bf(acc);
  }
}

// dst_bf[r,:] = bf16(src[idx[r],:])
__global__ void sel_rows_bf(const float* __restrict__ src, const int* __restrict__ idx,
                            u16* __restrict__ dst, int D) {
  int r = blockIdx.x, sr = idx[r];
  for (int d = threadIdx.x; d < D; d += blockDim.x)
    dst[(size_t)r * D + d] = f2bf(src[(size_t)sr * D + d]);
}

// per-chunk softmax over 4 members, chunkEmb[e,k,:] = sum_c w_c * Cu[4k+c,:]
__global__ void chunk_att(const float* __restrict__ sc, const float* __restrict__ Cu,
                          float* __restrict__ Y) {
  int ek = blockIdx.x, e = ek >> 6, k = ek & 63;
  const float* s = sc + (size_t)e * NC_ + k * 4;
  float m = fmaxf(fmaxf(s[0], s[1]), fmaxf(s[2], s[3]));
  float w0 = expf(s[0] - m), w1 = expf(s[1] - m), w2 = expf(s[2] - m), w3 = expf(s[3] - m);
  float inv = 1.f / (w0 + w1 + w2 + w3);
  const float* c0 = Cu + (size_t)(k * 4) * D_;
  for (int d = threadIdx.x; d < D_; d += 256) {
    float v = w0 * c0[d] + w1 * c0[D_ + d] + w2 * c0[2 * D_ + d] + w3 * c0[3 * D_ + d];
    Y[(size_t)ek * D_ + d] = v * inv;
  }
}

// delta2 row (padded to KP_): [E | Y | ||E-Y|| | E*Y | pos | 0-pad]
__global__ void build_d2(const float* __restrict__ E, const float* __restrict__ Y,
                         const float* __restrict__ pos, u16* __restrict__ dd) {
  __shared__ float red[256];
  int ek = blockIdx.x, e = ek >> 6;
  const float* Er = E + (size_t)e * D_;
  const float* Yr = Y + (size_t)ek * D_;
  float acc = 0.f;
  for (int d = threadIdx.x; d < D_; d += 256) { float df = Er[d] - Yr[d]; acc += df * df; }
  red[threadIdx.x] = acc; __syncthreads();
  for (int s = 128; s > 0; s >>= 1) {
    if (threadIdx.x < s) red[threadIdx.x] += red[threadIdx.x + s];
    __syncthreads();
  }
  float nrm = sqrtf(red[0]);
  u16* row = dd + (size_t)ek * KP_;
  for (int d = threadIdx.x; d < D_; d += 256) {
    row[d] = f2bf(Er[d]);
    row[D_ + d] = f2bf(Yr[d]);
    row[2 * D_ + 1 + d] = f2bf(Er[d] * Yr[d]);
  }
  if (threadIdx.x == 0) row[2 * D_] = f2bf(nrm);
  const float* pr = pos + (size_t)ek * POS_;
  for (int j = threadIdx.x; j < POS_; j += 256) row[3 * D_ + 1 + j] = f2bf(pr[j]);
  for (int j = threadIdx.x; j < KP_ - (3 * D_ + 1 + POS_); j += 256)
    row[3 * D_ + 1 + POS_ + j] = 0;
}

// delta3 row (padded): [E | Cu[n] | ||E-Cu|| | E*Cu | dis | extra | 0-pad]
__global__ void build_d3(const float* __restrict__ E, const float* __restrict__ Cu,
                         const float* __restrict__ dis, const float* __restrict__ feat,
                         u16* __restrict__ dd) {
  __shared__ float red[256];
  int en = blockIdx.x, e = en >> 8, n = en & 255;
  const float* Er = E + (size_t)e * D_;
  const float* Yr = Cu + (size_t)n * D_;
  float acc = 0.f;
  for (int d = threadIdx.x; d < D_; d += 256) { float df = Er[d] - Yr[d]; acc += df * df; }
  red[threadIdx.x] = acc; __syncthreads();
  for (int s = 128; s > 0; s >>= 1) {
    if (threadIdx.x < s) red[threadIdx.x] += red[threadIdx.x + s];
    __syncthreads();
  }
  float nrm = sqrtf(red[0]);
  u16* row = dd + (size_t)en * KP_;
  for (int d = threadIdx.x; d < D_; d += 256) {
    row[d] = f2bf(Er[d]);
    row[D_ + d] = f2bf(Yr[d]);
    row[2 * D_ + 1 + d] = f2bf(Er[d] * Yr[d]);
  }
  if (threadIdx.x == 0) {
    row[2 * D_] = f2bf(nrm);
    row[3 * D_ + 1 + POS_] = f2bf(feat[e * NK_ + (n >> 2)]);   // extra
  }
  const float* dr = dis + (size_t)n * POS_;
  for (int j = threadIdx.x; j < POS_; j += 256) row[3 * D_ + 1 + j] = f2bf(dr[j]);
  for (int j = threadIdx.x; j < KP_ - (3 * D_ + 2 + POS_); j += 256)
    row[3 * D_ + 2 + POS_ + j] = 0;
}

// BatchNorm over axis 1 of [OUTER, G, D] (biased var, eps 1e-5) + leaky relu
__global__ void bn_lrelu(const float* __restrict__ x, float* __restrict__ y,
                         int OUTER, int G, int D) {
  long long i = (long long)blockIdx.x * blockDim.x + threadIdx.x;
  if (i >= (long long)OUTER * D) return;
  int o = (int)(i / D), d = (int)(i % D);
  const float* bx = x + (size_t)o * G * D + d;
  float s = 0.f, sq = 0.f;
  for (int g = 0; g < G; ++g) { float v = bx[(size_t)g * D]; s += v; sq += v * v; }
  float m = s / G;
  float var = sq / G - m * m;
  if (var < 0.f) var = 0.f;
  float inv = rsqrtf(var + 1e-5f);
  float* by = y + (size_t)o * G * D + d;
  for (int g = 0; g < G; ++g) {
    float v = (bx[(size_t)g * D] - m) * inv;
    by[(size_t)g * D] = lrelu(v);
  }
}

// 2-class head: logits = H[row,:]@W[768,2] + b, out = log_softmax; feat = argmax
__global__ void head2(const float* __restrict__ H, const float* __restrict__ W,
                      const float* __restrict__ bvec, float* __restrict__ out,
                      float* __restrict__ feat) {
  __shared__ float r0[256], r1[256];
  size_t row = blockIdx.x;
  float a0 = 0.f, a1 = 0.f;
  for (int d = threadIdx.x; d < D_; d += 256) {
    float h = H[row * D_ + d];
    a0 += h * W[d * 2];
    a1 += h * W[d * 2 + 1];
  }
  r0[threadIdx.x] = a0; r1[threadIdx.x] = a1; __syncthreads();
  for (int s = 128; s > 0; s >>= 1) {
    if (threadIdx.x < s) { r0[threadIdx.x] += r0[threadIdx.x + s]; r1[threadIdx.x] += r1[threadIdx.x + s]; }
    __syncthreads();
  }
  if (threadIdx.x == 0) {
    float l0 = r0[0] + bvec[0], l1 = r1[0] + bvec[1];
    float mm = fmaxf(l0, l1);
    float lse = mm + logf(expf(l0 - mm) + expf(l1 - mm));
    out[row * 2] = l0 - lse;
    out[row * 2 + 1] = l1 - lse;
    if (feat) feat[row] = (l1 > l0) ? 1.f : 0.f;
  }
}

// L[e,n] = 1 if any valid label3[e,i] == n
__global__ void lmask(const int* __restrict__ lab, float* __restrict__ L) {
  int i = blockIdx.x * blockDim.x + threadIdx.x;
  if (i >= NE_ * NC_) return;
  int e = i / NC_, n = i % NC_;
  const int* l = lab + e * 3;
  L[i] = ((l[0] == n) || (l[1] == n) || (l[2] == n)) ? 1.f : 0.f;
}

// ---------------------------------------------------------------------------
// Host driver
// ---------------------------------------------------------------------------
extern "C" void kernel_launch(void* const* d_in, const int* in_sizes, int n_in,
                              void* d_out, int out_size, void* d_ws, size_t ws_size,
                              hipStream_t stream) {
  (void)in_sizes; (void)n_in; (void)out_size; (void)ws_size;
  const float* word = (const float*)d_in[0];
  const float* pos  = (const float*)d_in[1];
  const float* dis  = (const float*)d_in[2];
  const float* WihF = (const float*)d_in[3];
  const float* WhhF = (const float*)d_in[4];
  const float* bihF = (const float*)d_in[5];
  const float* bhhF = (const float*)d_in[6];
  const float* WihB = (const float*)d_in[7];
  const float* WhhB = (const float*)d_in[8];
  const float* bihB = (const float*)d_in[9];
  const float* bhhB = (const float*)d_in[10];
  const float* s1W  = (const float*)d_in[11];
  const float* s1b  = (const float*)d_in[12];
  const float* s2W  = (const float*)d_in[13];
  const float* repW = (const float*)d_in[14];
  const float* repb = (const float*)d_in[15];
  const float* W2W  = (const float*)d_in[16];
  const float* W2b  = (const float*)d_in[17];
  const float* WoW  = (const float*)d_in[18];
  const float* Wob  = (const float*)d_in[19];
  const float* W3W  = (const float*)d_in[20];
  const float* W3b  = (const float*)d_in[21];
  const float* clsW = (const float*)d_in[22];
  const float* clsb = (const float*)d_in[23];
  const int* slen   = (const int*)d_in[24];
  const int* emoI   = (const int*)d_in[25];
  const int* cauI   = (const int*)d_in[26];
  const int* lab    = (const int*)d_in[27];
  float* out = (float*)d_out;

  char* base = (char*)d_ws;
  size_t off = 0;
  auto alloc = [&](size_t b) -> char* {
    char* p = base + off;
    off = (off + b + 255) & ~(size_t)255;
    return p;
  };

  const size_t BT = (size_t)B_ * T_;
  u16* we_bf   = (u16*)alloc(BT * D_ * 2);
  u16* wihF_bf = (u16*)alloc((size_t)G4_ * D_ * 2);
  u16* whhF_bf = (u16*)alloc((size_t)G4_ * H_ * 2);
  u16* wihB_bf = (u16*)alloc((size_t)G4_ * D_ * 2);
  u16* whhB_bf = (u16*)alloc((size_t)G4_ * H_ * 2);
  u16* s1T_bf  = (u16*)alloc((size_t)D_ * D_ * 2);
  u16* repT_bf = (u16*)alloc((size_t)D_ * D_ * 2);
  u16* W2T_bf  = (u16*)alloc((size_t)D_ * KP_ * 2);
  u16* W3T_bf  = (u16*)alloc((size_t)D_ * KP_ * 2);
  float* bsumF = (float*)alloc(G4_ * 4);
  float* bsumB = (float*)alloc(G4_ * 4);
  // big region: LSTM gates [BT,4H] f32, later reused for delta3 bf16 [32768,KP_]
  size_t big_bytes = (size_t)NE_ * NC_ * KP_ * 2;   // 155 MB > gates' 151 MB
  char* big = alloc(big_bytes);
  float* gates = (float*)big;
  u16* delta3_bf = (u16*)big;
  float* gtmp  = (float*)alloc((size_t)B_ * G4_ * 4);
  u16*   hbf   = (u16*)alloc((size_t)B_ * H_ * 2);
  float* cbuf  = (float*)alloc((size_t)B_ * H_ * 4);
  float* Hs    = (float*)alloc(BT * D_ * 4);
  u16* xrev_bf = (u16*)alloc(BT * D_ * 2);
  float* hsbt  = (float*)alloc(BT * H_ * 4);
  u16* Hs_bf   = (u16*)alloc(BT * D_ * 2);
  float* S1    = (float*)alloc(BT * D_ * 4);
  float* ebuf  = (float*)alloc(BT * 4);
  float* U     = (float*)alloc((size_t)B_ * D_ * 4);
  u16* U_bf    = (u16*)alloc((size_t)B_ * D_ * 2);
  u16* Ue_bf   = (u16*)alloc((size_t)NE_ * D_ * 2);
  u16* Uc_bf   = (u16*)alloc((size_t)NC_ * D_ * 2);
  float* Ef    = (float*)alloc((size_t)NE_ * D_ * 4);
  float* Cu    = (float*)alloc((size_t)NC_ * D_ * 4);
  u16* E_bf    = (u16*)alloc((size_t)NE_ * D_ * 2);
  u16* Cu_bf   = (u16*)alloc((size_t)NC_ * D_ * 2);
  float* scrs  = (float*)alloc((size_t)NE_ * NC_ * 4);
  float* chY   = (float*)alloc((size_t)NE_ * NK_ * D_ * 4);
  u16* d2_bf   = (u16*)alloc((size_t)NE_ * NK_ * KP_ * 2);
  float* h2p   = (float*)alloc((size_t)NE_ * NK_ * D_ * 4);
  float* h2    = (float*)alloc((size_t)NE_ * NK_ * D_ * 4);
  float* feat  = (float*)alloc((size_t)NE_ * NK_ * 4);
  float* h3p   = (float*)alloc((size_t)NE_ * NC_ * D_ * 4);

  const int TB = 256;
  auto nb = [](size_t n) { return (unsigned)((n + 255) / 256); };

  // ---- stage 0: conversions ------------------------------------------------
  cvt_bf<<<nb(BT * D_), TB, 0, stream>>>(word, we_bf, BT * D_);
  cvt_bf<<<nb((size_t)G4_ * D_), TB, 0, stream>>>(WihF, wihF_bf, (size_t)G4_ * D_);
  cvt_bf<<<nb((size_t)G4_ * H_), TB, 0, stream>>>(WhhF, whhF_bf, (size_t)G4_ * H_);
  cvt_bf<<<nb((size_t)G4_ * D_), TB, 0, stream>>>(WihB, wihB_bf, (size_t)G4_ * D_);
  cvt_bf<<<nb((size_t)G4_ * H_), TB, 0, stream>>>(WhhB, whhB_bf, (size_t)G4_ * H_);
  cvt_tr_pad<<<nb((size_t)D_ * D_), TB, 0, stream>>>(s1W, s1T_bf, D_, D_, D_);
  cvt_tr_pad<<<nb((size_t)D_ * D_), TB, 0, stream>>>(repW, repT_bf, D_, D_, D_);
  cvt_tr_pad<<<nb((size_t)D_ * KP_), TB, 0, stream>>>(W2W, W2T_bf, 3 * D_ + 1 + POS_, D_, KP_);
  cvt_tr_pad<<<nb((size_t)D_ * KP_), TB, 0, stream>>>(W3W, W3T_bf, 3 * D_ + 2 + POS_, D_, KP_);
  add2<<<nb(G4_), TB, 0, stream>>>(bihF, bhhF, bsumF, G4_);
  add2<<<nb(G4_), TB, 0, stream>>>(bihB, bhhB, bsumB, G4_);

  // ---- forward LSTM --------------------------------------------------------
  wmma_gemm_2x2<<<dim3(G4_ / 128, (unsigned)(BT / 64)), TB, 0, stream>>>(
      we_bf, D_, wihF_bf, D_, bsumF, gates, G4_, (int)BT, G4_, D_, 0);
  hipMemsetAsync(hbf, 0, (size_t)B_ * H_ * 2, stream);
  hipMemsetAsync(cbuf, 0, (size_t)B_ * H_ * 4, stream);
  for (int t = 0; t < T_; ++t) {
    wmma_gemm<<<dim3(G4_ / 64, B_ / 32), TB, 0, stream>>>(
        hbf, H_, whhF_bf, H_, nullptr, gates + (size_t)t * G4_,
        (long long)T_ * G4_, gtmp, G4_, B_, G4_, H_, 0);
    lstm_pw<<<nb((size_t)B_ * H_), TB, 0, stream>>>(
        gtmp, cbuf, Hs + (size_t)t * D_, (size_t)T_ * D_, hbf);
  }

  // ---- backward LSTM -------------------------------------------------------
  gather_rev_bf<<<(unsigned)BT, TB, 0, stream>>>(we_bf, xrev_bf, slen);
  wmma_gemm_2x2<<<dim3(G4_ / 128, (unsigned)(BT / 64)), TB, 0, stream>>>(
      xrev_bf, D_, wihB_bf, D_, bsumB, gates, G4_, (int)BT, G4_, D_, 0);
  hipMemsetAsync(hbf, 0, (size_t)B_ * H_ * 2, stream);
  hipMemsetAsync(cbuf, 0, (size_t)B_ * H_ * 4, stream);
  for (int t = 0; t < T_; ++t) {
    wmma_gemm<<<dim3(G4_ / 64, B_ / 32), TB, 0, stream>>>(
        hbf, H_, whhB_bf, H_, nullptr, gates + (size_t)t * G4_,
        (long long)T_ * G4_, gtmp, G4_, B_, G4_, H_, 0);
    lstm_pw<<<nb((size_t)B_ * H_), TB, 0, stream>>>(
        gtmp, cbuf, hsbt + (size_t)t * H_, (size_t)T_ * H_, hbf);
  }
  gather_rev_h<<<(unsigned)BT, TB, 0, stream>>>(hsbt, Hs, slen);

  // ---- attention pooling ---------------------------------------------------
  cvt_bf<<<nb(BT * D_), TB, 0, stream>>>(Hs, Hs_bf, BT * D_);
  wmma_gemm_2x2<<<dim3(D_ / 128, (unsigned)(BT / 64)), TB, 0, stream>>>(
      Hs_bf, D_, s1T_bf, D_, s1b, S1, D_, (int)BT, D_, D_, 1);
  rowdot<<<(unsigned)BT, TB, 0, stream>>>(S1, s2W, ebuf, D_);
  attn_u<<<B_, TB, 0, stream>>>(ebuf, Hs, slen, U, U_bf);

  // ---- shared representations ---------------------------------------------
  sel_rows_bf<<<NE_, TB, 0, stream>>>(U, emoI, Ue_bf, D_);
  sel_rows_bf<<<NC_, TB, 0, stream>>>(U, cauI, Uc_bf, D_);
  wmma_gemm<<<dim3(D_ / 64, NE_ / 32), TB, 0, stream>>>(
      Ue_bf, D_, repT_bf, D_, repb, nullptr, 0, Ef, D_, NE_, D_, D_, 1);
  wmma_gemm<<<dim3(D_ / 64, NC_ / 32), TB, 0, stream>>>(
      Uc_bf, D_, repT_bf, D_, repb, nullptr, 0, Cu, D_, NC_, D_, D_, 1);
  cvt_bf<<<nb((size_t)NE_ * D_), TB, 0, stream>>>(Ef, E_bf, (size_t)NE_ * D_);
  cvt_bf<<<nb((size_t)NC_ * D_), TB, 0, stream>>>(Cu, Cu_bf, (size_t)NC_ * D_);

  // ---- chunk cross-attention ----------------------------------------------
  wmma_gemm<<<dim3(NC_ / 64, NE_ / 32), TB, 0, stream>>>(
      E_bf, D_, Cu_bf, D_, nullptr, nullptr, 0, scrs, NC_, NE_, NC_, D_, 0);
  chunk_att<<<NE_ * NK_, TB, 0, stream>>>(scrs, Cu, chY);

  // ---- phase 2 head --------------------------------------------------------
  build_d2<<<NE_ * NK_, TB, 0, stream>>>(Ef, chY, pos, d2_bf);
  wmma_gemm_2x2<<<dim3(D_ / 128, (NE_ * NK_) / 64), TB, 0, stream>>>(
      d2_bf, KP_, W2T_bf, KP_, W2b, h2p, D_, NE_ * NK_, D_, KP_, 0);
  bn_lrelu<<<nb((size_t)NE_ * D_), TB, 0, stream>>>(h2p, h2, NE_, NK_, D_);
  head2<<<NE_ * NK_, TB, 0, stream>>>(h2, WoW, Wob, out, feat);

  // ---- phase 3 head (reuses big region for delta3) -------------------------
  build_d3<<<NE_ * NC_, TB, 0, stream>>>(Ef, Cu, dis, feat, delta3_bf);
  wmma_gemm_2x2<<<dim3(D_ / 128, (NE_ * NC_) / 64), TB, 0, stream>>>(
      delta3_bf, KP_, W3T_bf, KP_, W3b, h3p, D_, NE_ * NC_, D_, KP_, 0);
  bn_lrelu<<<nb((size_t)NE_ * D_), TB, 0, stream>>>(h3p, h3p, NE_, NC_, D_);
  head2<<<NE_ * NC_, TB, 0, stream>>>(h3p, clsW, clsb, out + NE_ * NK_ * 2, nullptr);

  // ---- label mask ----------------------------------------------------------
  lmask<<<nb((size_t)NE_ * NC_), TB, 0, stream>>>(
      lab, out + NE_ * NK_ * 2 + NE_ * NC_ * 2);
  (void)cauI;
}
